// ShallowPrivateCollapsedDictNetSlow_7421703488055
// MI455X (gfx1250) — matrix operand order repaired and checked by reading, not compiled, not run on hardware
//
#include <hip/hip_runtime.h>

typedef __attribute__((ext_vector_type(16))) _Float16 v16h;
typedef __attribute__((ext_vector_type(8)))  _Float16 v8h;
typedef __attribute__((ext_vector_type(8)))  float    v8f;

#define NB 512
#define NC 22
#define TIN 1001
#define NK 40
#define KSZ 25
#define TOUT 976
#define NTILE 61          // 976 / 16
#define POOLW 100
#define NW 9
#define NOUTC 4
#define FCIN 360          // NK * NW
#define BN_EPS 1e-5f

// LDS layout
#define WT_HALVES (KSZ*48*32)    // 38400 halves = 76800 B : WT[j][k(48)][c(32)]
#define XT_HALVES (1000*32)      // 32000 halves = 64000 B : XT[t(1000)][c(32)]
#define WT_BYTES  (WT_HALVES*2)
#define XT_BYTES  (XT_HALVES*2)
#define ACC_FLOATS (6*16*11)     // per-wave: 16 k-slots x (9 pool + sum + sumsq)
#define LDS_BYTES (WT_BYTES + XT_BYTES + ACC_FLOATS*4)

// ---------------------------------------------------------------------------
// Kernel 1: per-sample conv (WMMA f16->f32) + bias + ELU + deterministic
// partial reductions for BN stats and pooled means.
// Grid: 512 blocks (one sample), 192 threads (6 waves, wave pair per M-tile).
// ---------------------------------------------------------------------------
__global__ __launch_bounds__(192) void conv_elu_pool_kernel(
    const float* __restrict__ x, const float* __restrict__ W,
    const float* __restrict__ bvec,
    float* __restrict__ Pool, float* __restrict__ Sum, float* __restrict__ Sq)
{
  extern __shared__ char smem[];
  _Float16* WT = (_Float16*)__builtin_assume_aligned(smem, 32);
  _Float16* XT = (_Float16*)__builtin_assume_aligned(smem + WT_BYTES, 32);
  float*   ACC = (float*)(smem + WT_BYTES + XT_BYTES);

  const int tid = threadIdx.x;
  const int b   = blockIdx.x;
  const size_t xoff = (size_t)b * NC * TIN;

  // subject id from marker x[b,0,1000] = sid * 1e6
  const int s = (int)(x[xoff + TIN - 1] * 1e-6f + 0.5f) - 1;

  for (int i = tid; i < ACC_FLOATS; i += 192) ACC[i] = 0.0f;

  // Stage weights for tap j: WT[(j*48 + k)*32 + c], zero-padded k>=40, c>=22
  for (int i = tid; i < WT_HALVES; i += 192) {
    const int c  = i & 31;
    const int kj = i >> 5;
    const int k  = kj % 48;
    const int j  = kj / 48;
    float v = 0.0f;
    if (c < NC && k < NK)
      v = W[(((size_t)s * NK + k) * NC + c) * KSZ + j];
    WT[i] = (_Float16)v;
  }
  // Stage x transposed: XT[t*32 + c]; coalesced global reads (t fast per lane)
  for (int i = tid; i < 32 * 1024; i += 192) {
    const int c = i >> 10;
    const int t = i & 1023;
    if (t < 1000) {
      const float v = (c < NC) ? x[xoff + (size_t)c * TIN + t] : 0.0f;
      XT[t * 32 + c] = (_Float16)v;
    }
  }
  __syncthreads();

  const int lane  = tid & 31;
  const int wv    = tid >> 5;           // 0..5
  const int mtile = wv >> 1;            // 0..2  (k rows 0-15 / 16-31 / 32-47)
  const int par   = wv & 1;             // ntile parity handled by this wave
  const int m16   = lane & 15;
  const int hi16  = (lane < 16) ? 0 : 1;
  const int kbA   = hi16 ? 8 : 0;       // A frag: K = kbA+0..7 and kbA+16..23
  const int kbB   = hi16 ? 8 : 0;       // B frag: K = kbB+0..7 and kbB+16..23
  const int kRow  = mtile * 16 + m16;

  float biasr[8];
  bool  kval[8];
#pragma unroll
  for (int r = 0; r < 8; ++r) {
    const int kg = mtile * 16 + r + (hi16 ? 8 : 0);
    kval[r]  = (kg < NK);
    biasr[r] = kval[r] ? bvec[s * NK + kg] : 0.0f;
  }

  float* accp = ACC + wv * 16 * 11;

  for (int ntile = par; ntile < NTILE; ntile += 2) {
    const int t0 = ntile * 16;
    v8f acc = {};
    for (int j = 0; j < KSZ; ++j) {
      // A (16x32 f16): per-lane two aligned 16B chunks of one weight row
      const _Float16* pa = WT + (((j * 48 + kRow) << 5) + kbA);
      const v8h alo = *(const v8h*)pa;
      const v8h ahi = *(const v8h*)(pa + 16);
      const v16h a = __builtin_shufflevector(alo, ahi,
          0,1,2,3,4,5,6,7,8,9,10,11,12,13,14,15);
      // B (32x16 f16): per-lane two aligned 16B chunks of one XT row
      const _Float16* pb = XT + (((t0 + m16 + j) << 5) + kbB);
      const v8h blo = *(const v8h*)pb;
      const v8h bhi = *(const v8h*)(pb + 16);
      const v16h bf = __builtin_shufflevector(blo, bhi,
          0,1,2,3,4,5,6,7,8,9,10,11,12,13,14,15);
      acc = __builtin_amdgcn_wmma_f32_16x16x32_f16(
          false, a, false, bf, (short)0, acc, false, false);
    }

    const int t  = t0 + m16;            // output time of this lane's column
    const int w0 = t0 / POOLW;
    const int tb = (w0 + 1) * POOLW;    // window boundary (<= 900)
#pragma unroll
    for (int r = 0; r < 8; ++r) {
      float v = acc[r] + biasr[r];
      v = (v > 0.0f) ? v : (__expf(v) - 1.0f);       // ELU
      float vb  = kval[r] ? v : 0.0f;                // BN sum
      float v2  = vb * vb;                           // BN sumsq
      float vlo = (kval[r] && t < tb) ? v : 0.0f;    // pool window w0
      float vhi = (kval[r] && t >= tb && t < 900) ? v : 0.0f; // window w0+1
#pragma unroll
      for (int m = 1; m < 16; m <<= 1) {
        vb  += __shfl_xor(vb,  m, 32);
        v2  += __shfl_xor(v2,  m, 32);
        vlo += __shfl_xor(vlo, m, 32);
        vhi += __shfl_xor(vhi, m, 32);
      }
      if (m16 == 0) {                   // lanes 0 and 16; disjoint addresses
        float* a2 = accp + (r + (hi16 ? 8 : 0)) * 11;
        a2[9]  += vb;
        a2[10] += v2;
        if (t0 < 900) {
          a2[w0] += vlo;
          if (w0 + 1 <= 8) a2[w0 + 1] += vhi;
        }
      }
    }
  }
  __syncthreads();

  // Combine the two waves per mtile (fixed order => deterministic) and emit
  for (int i = tid; i < NK * NW; i += 192) {
    const int k = i / NW, w = i % NW;
    const int m = k >> 4, k16 = k & 15;
    const float p = ACC[((2 * m) * 16 + k16) * 11 + w]
                  + ACC[((2 * m + 1) * 16 + k16) * 11 + w];
    Pool[((size_t)b * NK + k) * NW + w] = p * (1.0f / POOLW);
  }
  for (int i = tid; i < NK; i += 192) {
    const int m = i >> 4, k16 = i & 15;
    const float s1 = ACC[((2 * m) * 16 + k16) * 11 + 9]
                   + ACC[((2 * m + 1) * 16 + k16) * 11 + 9];
    const float s2 = ACC[((2 * m) * 16 + k16) * 11 + 10]
                   + ACC[((2 * m + 1) * 16 + k16) * 11 + 10];
    Sum[(size_t)b * NK + i] = s1;
    Sq [(size_t)b * NK + i] = s2;
  }
}

// ---------------------------------------------------------------------------
// Kernel 2: finalize BN stats; fold BN affine into FC weights/bias.
//   A2[n][kw] = fc_w[n][kw] * gamma'/rstd ; b2[n] = fc_b[n] + sum fc_w * shift
// ---------------------------------------------------------------------------
__global__ __launch_bounds__(128) void stats_fold_kernel(
    const float* __restrict__ Sum, const float* __restrict__ Sq,
    const float* __restrict__ gamma, const float* __restrict__ beta,
    const float* __restrict__ fcw, const float* __restrict__ fcb,
    float* __restrict__ A2, float* __restrict__ B2)
{
  __shared__ float gp[NK], sh[NK];
  const int tid = threadIdx.x;
  if (tid < NK) {
    float s1 = 0.0f, s2 = 0.0f;
    for (int bb = 0; bb < NB; ++bb) {
      s1 += Sum[bb * NK + tid];
      s2 += Sq [bb * NK + tid];
    }
    const float invN = 1.0f / (float)(NB * TOUT);
    const float mean = s1 * invN;
    const float var  = s2 * invN - mean * mean;     // biased, as jnp.var
    const float g    = gamma[tid] * rsqrtf(var + BN_EPS);
    gp[tid] = g;
    sh[tid] = beta[tid] - mean * g;
  }
  __syncthreads();
  for (int i = tid; i < NOUTC * FCIN; i += 128)
    A2[i] = fcw[i] * gp[(i % FCIN) / NW];
  if (tid < NOUTC) {
    float acc = fcb[tid];
    for (int i = 0; i < FCIN; ++i) acc += fcw[tid * FCIN + i] * sh[i / NW];
    B2[tid] = acc;
  }
}

// ---------------------------------------------------------------------------
// Kernel 3: out[b][n] = Pool[b][:] . A2[n][:] + b2[n]   (512 x 360 x 4)
// ---------------------------------------------------------------------------
__global__ __launch_bounds__(256) void fc_kernel(
    const float* __restrict__ Pool, const float* __restrict__ A2,
    const float* __restrict__ B2, float* __restrict__ out)
{
  const int b = blockIdx.x * 256 + threadIdx.x;
  if (b >= NB) return;
  float a0 = B2[0], a1 = B2[1], a2 = B2[2], a3 = B2[3];
  const float* p = Pool + (size_t)b * FCIN;
  for (int i = 0; i < FCIN; ++i) {
    const float v = p[i];
    a0 += v * A2[0 * FCIN + i];
    a1 += v * A2[1 * FCIN + i];
    a2 += v * A2[2 * FCIN + i];
    a3 += v * A2[3 * FCIN + i];
  }
  float* o = out + (size_t)b * NOUTC;
  o[0] = a0; o[1] = a1; o[2] = a2; o[3] = a3;
}

// ---------------------------------------------------------------------------
extern "C" void kernel_launch(void* const* d_in, const int* in_sizes, int n_in,
                              void* d_out, int out_size, void* d_ws, size_t ws_size,
                              hipStream_t stream) {
  const float* x     = (const float*)d_in[0];
  const float* W     = (const float*)d_in[1];
  const float* bvec  = (const float*)d_in[2];
  const float* gamma = (const float*)d_in[3];
  const float* beta  = (const float*)d_in[4];
  const float* fcw   = (const float*)d_in[5];
  const float* fcb   = (const float*)d_in[6];
  float* out = (float*)d_out;

  float* Pool = (float*)d_ws;                 // 512*360
  float* Sum  = Pool + (size_t)NB * FCIN;     // 512*40
  float* Sq   = Sum  + (size_t)NB * NK;       // 512*40
  float* A2   = Sq   + (size_t)NB * NK;       // 4*360
  float* B2   = A2   + (size_t)NOUTC * FCIN;  // 4

  hipLaunchKernelGGL(conv_elu_pool_kernel, dim3(NB), dim3(192), LDS_BYTES, stream,
                     x, W, bvec, Pool, Sum, Sq);
  hipLaunchKernelGGL(stats_fold_kernel, dim3(1), dim3(128), 0, stream,
                     Sum, Sq, gamma, beta, fcw, fcb, A2, B2);
  hipLaunchKernelGGL(fc_kernel, dim3(2), dim3(256), 0, stream,
                     Pool, A2, B2, out);
}
